// CenterHeadGroupSbnet_38671885533845
// MI455X (gfx1250) — matrix operand order repaired and checked by reading, not compiled, not run on hardware
//
#include <hip/hip_runtime.h>
#include <stdint.h>

typedef __attribute__((ext_vector_type(16))) __bf16 v16bf;
typedef __attribute__((ext_vector_type(8)))  float  v8f;

#define EPSF 1e-5f

struct __align__(4) bf2 { __bf16 x, y; };

// ---------------------------------------------------------------------------
// Fold BN(eval) into per-channel scale/shift
// ---------------------------------------------------------------------------
__global__ void bn_affine_kernel(const float* __restrict__ gamma,
                                 const float* __restrict__ beta,
                                 const float* __restrict__ mean,
                                 const float* __restrict__ var,
                                 float* __restrict__ scale,
                                 float* __restrict__ shift, int C) {
    int c = blockIdx.x * blockDim.x + threadIdx.x;
    if (c < C) {
        float sc = gamma[c] * rsqrtf(var[c] + EPSF);
        scale[c] = sc;
        shift[c] = beta[c] - mean[c] * sc;
    }
}

// One-time fp32 -> bf16 weight conversion (enables async B128 weight staging)
__global__ void cvt_bf16_kernel(const float* __restrict__ src,
                                __bf16* __restrict__ dst, int n) {
    int i = blockIdx.x * blockDim.x + threadIdx.x;
    if (i < n) dst[i] = (__bf16)src[i];
}

// ---------------------------------------------------------------------------
// Implicit-GEMM 3x3 conv via V_WMMA_F32_16X16X32_BF16.
//   M = Cout (mult of 64), K = Cin*9 (mult of 32), N = B*Hout*Wout.
// Block: 256 threads = 8 wave32, computes a 64(M) x 128(N) output tile.
// A-tile staged with global_load_async_to_lds_b128 (ASYNCcnt);
// B-tile im2col-gathered and written as packed 32-bit LDS stores.
// ---------------------------------------------------------------------------
#define LDK 40   // 32 K + 8 pad; row stride 80B (16B aligned)

template <typename TIN>
__global__ __launch_bounds__(256)
void wmma_conv3x3(const TIN* __restrict__ in, const __bf16* __restrict__ wbf,
                  __bf16* __restrict__ out,
                  int B, int Cin, int H, int W, int pad,
                  int Cout, int Hout, int Wout,
                  const float* __restrict__ out_scale,
                  const float* __restrict__ out_shift,
                  int out_relu)
{
    const int Kt  = Cin * 9;
    const int HWo = Hout * Wout;
    const int N   = B * HWo;

    __shared__ __bf16 lds_a[64 * LDK];    // [m][k]  weights tile
    __shared__ __bf16 lds_b[128 * LDK];   // [n][k]  im2col tile

    const int tid  = threadIdx.x;
    const int lane = tid & 31;
    const int wid  = tid >> 5;
    const int wm   = wid >> 1;            // 0..3  (M sub-tile)
    const int wn   = wid & 1;             // 0..1  (N half, 64 pixels each)
    const int half = lane >> 4;
    const int l16  = lane & 15;

    const int n0 = blockIdx.x * 128;
    const int m0 = blockIdx.y * 64;

    // ---- A async-fill coords: thread owns 16B of row m ----
    const int fa_m    = tid >> 2;                       // 0..63
    const int fa_part = tid & 3;                        // 0..3 (8 bf16 each)
    const uint32_t lds_a_off =
        (uint32_t)(uintptr_t)&lds_a[fa_m * LDK + fa_part * 8];
    const __bf16* wrow = wbf + (size_t)(m0 + fa_m) * Kt + fa_part * 8;

    // ---- B fill coords: thread owns pixel n and k-pairs ----
    const int fb_n  = tid & 127;                        // 0..127
    const int fb_kp = tid >> 7;                         // 0/1; kp = fb_kp+2i

    const int p = n0 + fb_n;
    const bool pvalid = (p < N);
    int pb = 0, poh = 0, pow_ = 0;
    if (pvalid) {
        pb = p / HWo;
        int rem = p - pb * HWo;
        poh  = rem / Wout;
        pow_ = rem - poh * Wout;
    }

    v8f acc[4];
    v8f zero = {};
    #pragma unroll
    for (int j = 0; j < 4; ++j) acc[j] = zero;

    for (int k0 = 0; k0 < Kt; k0 += 32) {
        // ---- stage A: one async 16B global->LDS copy per thread ----
        {
            uint64_t ga = (uint64_t)(uintptr_t)(wrow + k0);
            asm volatile("global_load_async_to_lds_b128 %0, %1, off"
                         :: "v"(lds_a_off), "v"(ga) : "memory");
        }
        // ---- stage B: im2col gather, packed 32-bit LDS stores ----
        #pragma unroll
        for (int i = 0; i < 8; ++i) {
            int kp = fb_kp + 2 * i;       // pair index 0..15
            int k  = kp * 2;
            float v0 = 0.f, v1 = 0.f;
            if (pvalid) {
                int kk = k0 + k;
                {
                    int ci = kk / 9;
                    int t  = kk - ci * 9;
                    int kh = t / 3, kw = t - kh * 3;
                    int ih = poh + kh - pad, iw = pow_ + kw - pad;
                    if (ih >= 0 && ih < H && iw >= 0 && iw < W)
                        v0 = (float) in[((pb * Cin + ci) * H + ih) * W + iw];
                }
                {
                    int kk1 = kk + 1;
                    int ci = kk1 / 9;
                    int t  = kk1 - ci * 9;
                    int kh = t / 3, kw = t - kh * 3;
                    int ih = poh + kh - pad, iw = pow_ + kw - pad;
                    if (ih >= 0 && ih < H && iw >= 0 && iw < W)
                        v1 = (float) in[((pb * Cin + ci) * H + ih) * W + iw];
                }
            }
            bf2 pk;
            pk.x = (__bf16)v0;
            pk.y = (__bf16)v1;
            *(bf2*)&lds_b[fb_n * LDK + k] = pk;
        }
        // async A copies must land in LDS before anyone reads past barrier
        asm volatile("s_wait_asynccnt 0" ::: "memory");
        __syncthreads();

        // ---- A fragment (ISA 7.12.2: 16-bit A 16x32) ----
        v16bf afrag;
        {
            const __bf16* ar = &lds_a[(wm * 16 + l16) * LDK];
            #pragma unroll
            for (int e = 0; e < 16; ++e) {
                int v = e >> 1, s = e & 1;
                int k = ((v < 4) ? 0 : 16) + (half << 3) + ((v & 3) << 1) + s;
                afrag[e] = ar[k];
            }
        }
        // ---- 4 WMMAs over the wave's 64-pixel strip ----
        #pragma unroll
        for (int j = 0; j < 4; ++j) {
            v16bf bfrag;
            const __bf16* br =
                &lds_b[(wn * 64 + j * 16 + l16) * LDK + (half << 3)];
            #pragma unroll
            for (int e = 0; e < 16; ++e) bfrag[e] = br[((e >> 3) << 4) + (e & 7)];
            acc[j] = __builtin_amdgcn_wmma_f32_16x16x32_bf16(
                false, afrag, false, bfrag, (short)0, acc[j], false, false);
        }
        __syncthreads();
    }

    // ---- epilogue: optional BN affine + relu, store bf16 NCHW ----
    #pragma unroll
    for (int j = 0; j < 4; ++j) {
        int pp = n0 + wn * 64 + j * 16 + l16;
        if (pp < N) {
            int b  = pp / HWo;
            int r2 = pp - b * HWo;
            int oh = r2 / Wout;
            int ow = r2 - oh * Wout;
            #pragma unroll
            for (int r = 0; r < 8; ++r) {
                int m = m0 + wm * 16 + (half << 3) + r;
                float v = acc[j][r];
                if (out_scale) v = v * out_scale[m] + out_shift[m];
                if (out_relu)  v = fmaxf(v, 0.f);
                out[((b * Cout + m) * Hout + oh) * Wout + ow] = (__bf16) v;
            }
        }
    }
}

// ---------------------------------------------------------------------------
// GroupNorm stats -> per-(batch,channel) scale/shift with gamma/beta folded.
// ---------------------------------------------------------------------------
__global__ __launch_bounds__(1024)
void gn_stats_kernel(const __bf16* __restrict__ mid,
                     const float* __restrict__ gamma,
                     const float* __restrict__ beta,
                     float* __restrict__ scale, float* __restrict__ shift,
                     int C, int G, int HW)
{
    int bg = blockIdx.x;
    int b  = bg / G, g = bg - b * G;
    int Cg = C / G;
    long base = (long)(b * C + g * Cg) * HW;
    long n    = (long)Cg * HW;

    float s = 0.f, s2 = 0.f;
    for (long i = threadIdx.x; i < n; i += 1024) {
        float v = (float) mid[base + i];
        s  += v;
        s2 += v * v;
    }
    __shared__ float sh_s[1024], sh_s2[1024];
    sh_s[threadIdx.x]  = s;
    sh_s2[threadIdx.x] = s2;
    __syncthreads();
    for (int st = 512; st > 0; st >>= 1) {
        if (threadIdx.x < st) {
            sh_s[threadIdx.x]  += sh_s[threadIdx.x + st];
            sh_s2[threadIdx.x] += sh_s2[threadIdx.x + st];
        }
        __syncthreads();
    }
    float mean = sh_s[0] / (float)n;
    float var  = sh_s2[0] / (float)n - mean * mean;
    float inv  = rsqrtf(var + EPSF);
    if ((int)threadIdx.x < Cg) {
        int c = g * Cg + threadIdx.x;
        float sc = gamma[c] * inv;
        scale[b * C + c] = sc;
        shift[b * C + c] = beta[c] - mean * sc;
    }
}

// ---------------------------------------------------------------------------
// Small grouped 3x3 convs (direct VALU), GN affine+relu applied on load.
// ---------------------------------------------------------------------------
__global__ void hm1_kernel(const __bf16* __restrict__ mid,
                           const float* __restrict__ w,
                           const float* __restrict__ bias,
                           const float* __restrict__ scale,
                           const float* __restrict__ shift,
                           float* __restrict__ out)
{
    const int Hm = 188, Wm = 188;
    int idx = blockIdx.x * blockDim.x + threadIdx.x;
    int total = 2 * 12 * Hm * Wm;
    if (idx >= total) return;
    int ow = idx % Wm;  int t = idx / Wm;
    int oh = t % Hm;    t /= Hm;
    int oc = t % 12;    int b = t / 12;
    int cin0 = (oc >> 1) * 64;

    float acc = bias[oc];
    for (int ci = 0; ci < 64; ++ci) {
        int c = cin0 + ci;
        float sc = scale[b * 384 + c], sf = shift[b * 384 + c];
        const __bf16* mrow = mid + ((long)(b * 384 + c) * Hm) * Wm;
        const float*  wrow = w + (oc * 64 + ci) * 9;
        #pragma unroll
        for (int kh = 0; kh < 3; ++kh) {
            int ih = oh + kh - 1;
            if (ih < 0 || ih >= Hm) continue;
            #pragma unroll
            for (int kw = 0; kw < 3; ++kw) {
                int iw = ow + kw - 1;
                if (iw < 0 || iw >= Wm) continue;
                float v = (float) mrow[ih * Wm + iw];
                v = fmaxf(v * sc + sf, 0.f);
                acc += v * wrow[kh * 3 + kw];
            }
        }
    }
    out[idx] = acc;
}

__global__ void det1_kernel(const __bf16* __restrict__ mid,
                            const float* __restrict__ w,
                            const float* __restrict__ bias,
                            const float* __restrict__ scale,
                            const float* __restrict__ shift,
                            float* __restrict__ out)
{
    const int Hm = 186, Wm = 186, Ho = 184, Wo = 184;
    int idx = blockIdx.x * blockDim.x + threadIdx.x;
    int total = 2 * 15 * Ho * Wo;
    if (idx >= total) return;
    int ow = idx % Wo;  int t = idx / Wo;
    int oh = t % Ho;    t /= Ho;
    int oc = t % 15;    int b = t / 15;
    int cin0 = (oc / 3) * 64;

    float acc = bias[oc];
    for (int ci = 0; ci < 64; ++ci) {
        int c = cin0 + ci;
        float sc = scale[b * 320 + c], sf = shift[b * 320 + c];
        const __bf16* mrow = mid + ((long)(b * 320 + c) * Hm) * Wm;
        const float*  wrow = w + (oc * 64 + ci) * 9;
        #pragma unroll
        for (int kh = 0; kh < 3; ++kh) {
            int ih = oh + kh;
            #pragma unroll
            for (int kw = 0; kw < 3; ++kw) {
                float v = (float) mrow[ih * Wm + ow + kw];
                v = fmaxf(v * sc + sf, 0.f);
                acc += v * wrow[kh * 3 + kw];
            }
        }
    }
    out[idx] = acc;
}

// ---------------------------------------------------------------------------
// Orchestration
// ---------------------------------------------------------------------------
extern "C" void kernel_launch(void* const* d_in, const int* in_sizes, int n_in,
                              void* d_out, int out_size, void* d_ws, size_t ws_size,
                              hipStream_t stream) {
    (void)in_sizes; (void)n_in; (void)out_size; (void)ws_size;

    const float* x        = (const float*)d_in[0];
    const float* sc_w     = (const float*)d_in[1];
    const float* sc_gamma = (const float*)d_in[2];
    const float* sc_beta  = (const float*)d_in[3];
    const float* sc_mean  = (const float*)d_in[4];
    const float* sc_var   = (const float*)d_in[5];
    const float* hm_w0    = (const float*)d_in[6];
    const float* hm_gn_g  = (const float*)d_in[7];
    const float* hm_gn_b  = (const float*)d_in[8];
    const float* hm_w1    = (const float*)d_in[9];
    const float* hm_b1    = (const float*)d_in[10];
    const float* dh_w0    = (const float*)d_in[11];
    const float* dh_gn_g  = (const float*)d_in[12];
    const float* dh_gn_b  = (const float*)d_in[13];
    const float* dh_w1    = (const float*)d_in[14];
    const float* dh_b1    = (const float*)d_in[15];
    float* outf = (float*)d_out;

    const int B = 2, H = 188, W = 188, HW = H * W;          // 35344
    const int Hm = 186, Wm = 186, HWm = Hm * Wm;            // 34596
    const int N1 = B * HW;                                  // 70688
    const int N0 = B * HWm;                                 // 69192
    const int HM_OUT = B * 12 * HW;                         // 848256
    const int DET_PER_HEAD = B * 15 * 184 * 184;            // 1015680
    const int NW_SC  = 64 * 512 * 9;                        // 294912
    const int NW_HM0 = 384 * 64 * 9;                        // 221184
    const int NW_DH0 = 6 * 320 * 64 * 9;                    // 1105920

    // ---- workspace carve-up ----
    char* wp = (char*)d_ws;
    auto alloc = [&](size_t bytes) -> void* {
        void* r = (void*)wp;
        wp += (bytes + 255) & ~(size_t)255;
        return r;
    };
    __bf16* y        = (__bf16*)alloc((size_t)B * 64  * HW  * 2);
    __bf16* hm_mid   = (__bf16*)alloc((size_t)B * 384 * HW  * 2);
    __bf16* det_mid  = (__bf16*)alloc((size_t)B * 320 * HWm * 2);
    __bf16* w_sc_bf  = (__bf16*)alloc((size_t)NW_SC  * 2);
    __bf16* w_hm0_bf = (__bf16*)alloc((size_t)NW_HM0 * 2);
    __bf16* w_dh0_bf = (__bf16*)alloc((size_t)NW_DH0 * 2);
    float*  bn_scale = (float*) alloc(64 * 4);
    float*  bn_shift = (float*) alloc(64 * 4);
    float*  gn_hm_s  = (float*) alloc((size_t)B * 384 * 4);
    float*  gn_hm_f  = (float*) alloc((size_t)B * 384 * 4);
    float*  gn_dh_s  = (float*) alloc((size_t)B * 320 * 4);
    float*  gn_dh_f  = (float*) alloc((size_t)B * 320 * 4);

    // 0) one-time weight conversions to bf16 (feeds async B128 staging)
    cvt_bf16_kernel<<<(NW_SC  + 255) / 256, 256, 0, stream>>>(sc_w,  w_sc_bf,  NW_SC);
    cvt_bf16_kernel<<<(NW_HM0 + 255) / 256, 256, 0, stream>>>(hm_w0, w_hm0_bf, NW_HM0);
    cvt_bf16_kernel<<<(NW_DH0 + 255) / 256, 256, 0, stream>>>(dh_w0, w_dh0_bf, NW_DH0);

    // 1) fold BN into per-channel affine
    bn_affine_kernel<<<1, 64, 0, stream>>>(sc_gamma, sc_beta, sc_mean, sc_var,
                                           bn_scale, bn_shift, 64);

    // 2) shared_conv: 512->64, pad1, fused BN+ReLU, bf16 out   (WMMA)
    {
        dim3 grid((N1 + 127) / 128, 64 / 64);
        wmma_conv3x3<float><<<grid, 256, 0, stream>>>(
            x, w_sc_bf, y, B, 512, H, W, 1, 64, H, W,
            bn_scale, bn_shift, 1);
    }

    // 3) hm conv0: 64->384, pad1, raw bf16 out                  (WMMA)
    {
        dim3 grid((N1 + 127) / 128, 384 / 64);
        wmma_conv3x3<__bf16><<<grid, 256, 0, stream>>>(
            y, w_hm0_bf, hm_mid, B, 64, H, W, 1, 384, H, W,
            nullptr, nullptr, 0);
    }

    // 4) GN(6) stats -> per-(b,c) affine
    gn_stats_kernel<<<B * 6, 1024, 0, stream>>>(hm_mid, hm_gn_g, hm_gn_b,
                                                gn_hm_s, gn_hm_f, 384, 6, HW);

    // 5) grouped conv 384->12 + bias -> d_out[0 : 848256]
    hm1_kernel<<<(HM_OUT + 255) / 256, 256, 0, stream>>>(
        hm_mid, hm_w1, hm_b1, gn_hm_s, gn_hm_f, outf);

    // 6) six detection heads
    for (int h = 0; h < 6; ++h) {
        dim3 grid((N0 + 127) / 128, 320 / 64);
        wmma_conv3x3<__bf16><<<grid, 256, 0, stream>>>(
            y, w_dh0_bf + (size_t)h * 320 * 64 * 9, det_mid,
            B, 64, H, W, 0, 320, Hm, Wm,
            nullptr, nullptr, 0);

        gn_stats_kernel<<<B * 5, 1024, 0, stream>>>(
            det_mid, dh_gn_g + h * 320, dh_gn_b + h * 320,
            gn_dh_s, gn_dh_f, 320, 5, HWm);

        det1_kernel<<<(DET_PER_HEAD + 255) / 256, 256, 0, stream>>>(
            det_mid, dh_w1 + (size_t)h * 15 * 64 * 9, dh_b1 + h * 15,
            gn_dh_s, gn_dh_f, outf + HM_OUT + (size_t)h * DET_PER_HEAD);
    }
}
